// VaeDecoder_84361747628757
// MI455X (gfx1250) — matrix operand order repaired and checked by reading, not compiled, not run on hardware
//
#include <hip/hip_runtime.h>
#include <hip/hip_bf16.h>
#include <cstdint>

typedef _Float16 f16t;
typedef __attribute__((ext_vector_type(16))) _Float16 v16h;
typedef __attribute__((ext_vector_type(8)))  float    v8f;

namespace {
constexpr int   kB      = 262144;
constexpr int   kLatent = 100;
constexpr int   kEmb    = 64;
constexpr int   kH      = 64;
constexpr int   kGH     = 256;   // 4*H
constexpr int   kSeq    = 30;
constexpr float kDT     = 0.03f;
constexpr int   kWaves  = 8;     // waves per workgroup
constexpr int   kRowsWG = 128;   // 16 rows per wave * 8 waves
constexpr int   kPitch  = 136;   // f16 elements per A-panel row (128 + 8 pad)
}

union AFrag { v16h h; uint32_t u[8]; uint4 q[2]; };
union HPack { _Float16 h[2]; uint32_t u; };

struct SMem {
  _Float16 frag[64 * 512];              // B fragments, fragment-layout order (64 KB)
  _Float16 A[kWaves][16 * kPitch];      // per-wave A panel: cols 0..63 = x, 64..127 = h
  float W_emb[kEmb * 4];
  float b_emb[kEmb];
  float b_gates[kGH];                   // b_ih + b_hh
  float b_init[kH];
  float W_ctrl[2 * kH];
  float b_ctrl[2];
  float ctrl[kWaves][32];               // [0..15]=pedal, [16..31]=steering per row
  float state[kWaves][16][4];
};

__device__ __forceinline__ float sigf(float x) { return 1.f / (1.f + __expf(-x)); }
__device__ __forceinline__ float tanhfast(float x) {
  x = fminf(fmaxf(x, -15.f), 15.f);
  float e = __expf(2.f * x);
  return (e - 1.f) / (e + 1.f);
}

__global__ __launch_bounds__(256) void lstm_plant_rollout(
    const float* __restrict__ z, const float* __restrict__ init_state,
    const float* __restrict__ W_emb, const float* __restrict__ b_emb,
    const float* __restrict__ W_ih, const float* __restrict__ W_hh,
    const float* __restrict__ b_ih, const float* __restrict__ b_hh,
    const float* __restrict__ W_ctrl, const float* __restrict__ b_ctrl,
    const float* __restrict__ W_init, const float* __restrict__ b_init,
    float* __restrict__ out)
{
  __shared__ SMem sm;
  const int tid  = threadIdx.x;
  const int wv   = tid >> 5;
  const int lane = tid & 31;
  const int ln15 = lane & 15;
  const int lhi  = lane >> 4;
  const int rowbase = blockIdx.x * kRowsWG + wv * 16;

  // ---------------- phase 0: stage small weights + W_init B-fragments ----------------
  // B fragment layout (16-bit, 32x16 KxN): lane L holds column N=L%16,
  // element e -> k = 8*(L/16) + (e%8) + 16*(e/8).  Stored 512 f16 per fragment.
  for (int i = tid; i < 16 * 512; i += 256) {
    int f = i >> 9, r = i & 511, L = r >> 4, e = r & 15;
    int kc = f >> 2, nt = f & 3;
    int k = 32 * kc + 8 * (L >> 4) + (e & 7) + 16 * (e >> 3);
    int n = 16 * nt + (L & 15);
    float v = (k < kLatent) ? W_init[n * kLatent + k] : 0.f;
    sm.frag[i] = (f16t)v;
  }
  for (int i = tid; i < kEmb * 4; i += 256) sm.W_emb[i] = W_emb[i];
  for (int i = tid; i < kEmb;     i += 256) sm.b_emb[i] = b_emb[i];
  for (int i = tid; i < kGH;      i += 256) sm.b_gates[i] = b_ih[i] + b_hh[i];
  for (int i = tid; i < kH;       i += 256) sm.b_init[i] = b_init[i];
  for (int i = tid; i < 2 * kH;   i += 256) sm.W_ctrl[i] = W_ctrl[i];
  if (tid < 2) sm.b_ctrl[tid] = b_ctrl[tid];

  // z tile -> A panel (f16, K padded 100 -> 128)
  {
    int m = ln15;
    int kbase = 64 * lhi;
    const float* zr = z + (size_t)(rowbase + m) * kLatent;
    for (int kk = 0; kk < 64; ++kk) {
      int k = kbase + kk;
      float v = (k < kLatent) ? zr[k] : 0.f;
      sm.A[wv][m * kPitch + k] = (f16t)v;
    }
  }
  // initial plant state: lane r<16 owns row r of this wave's tile in registers
  float sx = 0.f, sy = 0.f, spsi = 0.f, svt = 0.f;
  if (lane < 16) {
    const float* ip = init_state + (size_t)(rowbase + ln15) * 4;
    sx = ip[0]; sy = ip[1]; spsi = ip[2]; svt = ip[3];
    sm.state[wv][ln15][0] = sx; sm.state[wv][ln15][1] = sy;
    sm.state[wv][ln15][2] = spsi; sm.state[wv][ln15][3] = svt;
  }
  __syncthreads();

  // ---------------- h0 = z @ W_init^T + b_init  (WMMA) ----------------
  // A fragment layout (16-bit 16x32 MxK): lane L row M=L%16,
  // element (v, half) -> k = 2v + half + 8*((v>=4) + (L>=16)).
  float c_reg[4][8];   // persistent cell state: c[m=v+8*lhi][j=16*tt+ln15]
  {
    AFrag a[4];
#pragma unroll
    for (int kc = 0; kc < 4; ++kc)
#pragma unroll
      for (int v = 0; v < 8; ++v) {
        int k = kc * 32 + 2 * v + 8 * ((v >> 2) + lhi);
        a[kc].u[v] = *(const uint32_t*)&sm.A[wv][ln15 * kPitch + k];
      }
#pragma unroll
    for (int nt = 0; nt < 4; ++nt) {
      float bv = sm.b_init[16 * nt + ln15];
      v8f acc;
#pragma unroll
      for (int v = 0; v < 8; ++v) acc[v] = bv;
#pragma unroll
      for (int kc = 0; kc < 4; ++kc) {
        AFrag b;
        const uint4* bp = (const uint4*)&sm.frag[(kc * 4 + nt) * 512 + lane * 16];
        b.q[0] = bp[0]; b.q[1] = bp[1];
        acc = __builtin_amdgcn_wmma_f32_16x16x32_f16(false, a[kc].h, false, b.h,
                                                     (short)0, acc, false, false);
      }
#pragma unroll
      for (int v = 0; v < 8; ++v) {
        float h0 = acc[v];
        c_reg[nt][v] = h0;                       // c0 = h0
        int m = v + 8 * lhi;
        sm.A[wv][m * kPitch + 64 + 16 * nt + ln15] = (f16t)h0;
      }
    }
  }
  __syncthreads();

  // ---------------- refill fragment region with gate weights [W_ih ; W_hh] ----------------
  for (int i = tid; i < 64 * 512; i += 256) {
    int f = i >> 9, r = i & 511, L = r >> 4, e = r & 15;
    int kc = f >> 4, nt = f & 15;
    int k = 32 * kc + 8 * (L >> 4) + (e & 7) + 16 * (e >> 3);
    int n = 16 * nt + (L & 15);
    float v = (k < 64) ? W_ih[n * kEmb + k] : W_hh[n * kH + (k - 64)];
    sm.frag[i] = (f16t)v;
  }
  __syncthreads();

  // ---------------- 30-step rollout ----------------
  for (int t = 0; t < kSeq; ++t) {
    // x = prev_state @ W_emb^T + b_emb  -> A cols 0..63 (f16)
    {
      int m = ln15;
      int kbase = 32 * lhi;
      float s0 = sm.state[wv][m][0], s1 = sm.state[wv][m][1];
      float s2 = sm.state[wv][m][2], s3 = sm.state[wv][m][3];
#pragma unroll
      for (int kk = 0; kk < 32; kk += 2) {
        int k = kbase + kk;
        float x0 = sm.b_emb[k] + s0 * sm.W_emb[k * 4 + 0] + s1 * sm.W_emb[k * 4 + 1]
                               + s2 * sm.W_emb[k * 4 + 2] + s3 * sm.W_emb[k * 4 + 3];
        float x1 = sm.b_emb[k + 1] + s0 * sm.W_emb[(k + 1) * 4 + 0] + s1 * sm.W_emb[(k + 1) * 4 + 1]
                                   + s2 * sm.W_emb[(k + 1) * 4 + 2] + s3 * sm.W_emb[(k + 1) * 4 + 3];
        HPack p; p.h[0] = (f16t)x0; p.h[1] = (f16t)x1;
        *(uint32_t*)&sm.A[wv][m * kPitch + k] = p.u;
      }
    }
    __syncthreads();

    // A fragments for [x|h] (K = 128)
    AFrag a[4];
#pragma unroll
    for (int kc = 0; kc < 4; ++kc)
#pragma unroll
      for (int v = 0; v < 8; ++v) {
        int k = kc * 32 + 2 * v + 8 * ((v >> 2) + lhi);
        a[kc].u[v] = *(const uint32_t*)&sm.A[wv][ln15 * kPitch + k];
      }

    // gates = [x|h] @ [W_ih;W_hh]^T + b ; gate tiles tt, tt+4, tt+8, tt+12 = i,f,g,o
#pragma unroll
    for (int tt = 0; tt < 4; ++tt) {
      float bi = sm.b_gates[      16 * tt + ln15];
      float bf = sm.b_gates[ 64 + 16 * tt + ln15];
      float bg = sm.b_gates[128 + 16 * tt + ln15];
      float bo = sm.b_gates[192 + 16 * tt + ln15];
      v8f accs[4];
#pragma unroll
      for (int v = 0; v < 8; ++v) {
        accs[0][v] = bi; accs[1][v] = bf; accs[2][v] = bg; accs[3][v] = bo;
      }
#pragma unroll
      for (int kc = 0; kc < 4; ++kc) {
#pragma unroll
        for (int q = 0; q < 4; ++q) {
          int nt = tt + 4 * q;
          AFrag b;
          const uint4* bp = (const uint4*)&sm.frag[((kc << 4) + nt) * 512 + lane * 16];
          b.q[0] = bp[0]; b.q[1] = bp[1];
          accs[q] = __builtin_amdgcn_wmma_f32_16x16x32_f16(false, a[kc].h, false, b.h,
                                                           (short)0, accs[q], false, false);
        }
      }
      // lane-local LSTM cell update (same (m,j) across the four gate tiles)
#pragma unroll
      for (int v = 0; v < 8; ++v) {
        float ig = sigf(accs[0][v]);
        float fg = sigf(accs[1][v]);
        float gg = tanhfast(accs[2][v]);
        float og = sigf(accs[3][v]);
        float c  = fg * c_reg[tt][v] + ig * gg;
        c_reg[tt][v] = c;
        float h = og * tanhfast(c);
        int m = v + 8 * lhi;
        sm.A[wv][m * kPitch + 64 + 16 * tt + ln15] = (f16t)h;
      }
    }
    __syncthreads();

    // control head: lane L -> row r=L%16, channel cc=L/16 (0=pedal, 1=steering)
    {
      int r = ln15, cc = lhi;
      float acc = sm.b_ctrl[cc];
      const _Float16* hr = &sm.A[wv][r * kPitch + 64];
      const float* wc = &sm.W_ctrl[cc * kH];
#pragma unroll
      for (int j = 0; j < 64; j += 2) {
        HPack p; p.u = *(const uint32_t*)&hr[j];
        acc += (float)p.h[0] * wc[j] + (float)p.h[1] * wc[j + 1];
      }
      sm.ctrl[wv][cc * 16 + r] = acc;
    }
    __syncthreads();

    // plant dynamics + trajectory output (lanes 0..15 own one row each)
    if (lane < 16) {
      float pedal = sm.ctrl[wv][ln15];
      float steer = sm.ctrl[wv][16 + ln15];
      float beta  = fminf(fmaxf(steer, -0.5f), 0.5f);
      float v1    = fminf(fmaxf(svt + pedal * kDT, 0.f), 10.f);
      float xd    = v1 * __cosf(spsi);
      float yd    = v1 * __sinf(spsi);
      float psid  = fminf(fmaxf(svt * __tanf(beta) * (1.f / 2.5f), -1.57f), 1.57f);
      sx += xd * kDT; sy += yd * kDT; spsi += psid * kDT; svt = v1;
      sm.state[wv][ln15][0] = sx; sm.state[wv][ln15][1] = sy;
      sm.state[wv][ln15][2] = spsi; sm.state[wv][ln15][3] = svt;
      float* op = out + ((size_t)(rowbase + ln15) * kSeq + t) * 4;
      op[0] = sx; op[1] = sy; op[2] = spsi; op[3] = svt;
    }
    __syncthreads();
  }
}

extern "C" void kernel_launch(void* const* d_in, const int* in_sizes, int n_in,
                              void* d_out, int out_size, void* d_ws, size_t ws_size,
                              hipStream_t stream) {
  (void)in_sizes; (void)n_in; (void)out_size; (void)d_ws; (void)ws_size;
  const float* z          = (const float*)d_in[0];
  const float* init_state = (const float*)d_in[1];
  const float* W_emb      = (const float*)d_in[2];
  const float* b_emb      = (const float*)d_in[3];
  const float* W_ih       = (const float*)d_in[4];
  const float* W_hh       = (const float*)d_in[5];
  const float* b_ih       = (const float*)d_in[6];
  const float* b_hh       = (const float*)d_in[7];
  const float* W_ctrl     = (const float*)d_in[8];
  const float* b_ctrl     = (const float*)d_in[9];
  const float* W_init     = (const float*)d_in[10];
  const float* b_init     = (const float*)d_in[11];
  float* out = (float*)d_out;

  dim3 grid(kB / kRowsWG);   // 2048 workgroups of 256 threads (8 wave32)
  lstm_plant_rollout<<<grid, 256, 0, stream>>>(z, init_state, W_emb, b_emb, W_ih, W_hh,
                                               b_ih, b_hh, W_ctrl, b_ctrl, W_init, b_init,
                                               out);
}
